// GroupedQueryAttention_36163624632989
// MI455X (gfx1250) — compile-verified, matrix operand order checked by reading, not run on hardware
//
#include <hip/hip_runtime.h>
#include <hip/hip_bf16.h>
#include <math.h>

typedef __attribute__((ext_vector_type(16))) _Float16 v16h;
typedef __attribute__((ext_vector_type(8)))  _Float16 v8h;
typedef __attribute__((ext_vector_type(4)))  _Float16 v4h;
typedef __attribute__((ext_vector_type(8)))  float    v8f;

#define BDIM 2
#define SEQ  2048
#define EDIM 2048
#define NQH  32
#define NKVH 8
#define HD   64
#define KVE  512   // NKVH * HD

// log2(e)/sqrt(64), pre-folded into the Q->f16 conversion
#define QSCALE 0.1803368801111204f

// ---- helpers -------------------------------------------------------------

constexpr int dpp8sel(int a,int b,int c,int d,int e,int f,int g,int h) {
  return a | (b<<3) | (c<<6) | (d<<9) | (e<<12) | (f<<15) | (g<<18) | (h<<21);
}

// lane-xor shuffles done on the VALU (no LDS traffic):
// xor1/2/4 via DPP8, xor8 via DPP16 ROW_ROR:8 (= xor8 within a 16-lane row)
__device__ __forceinline__ float dpp_xor1(float x) {
  return __int_as_float(__builtin_amdgcn_mov_dpp8(
      __float_as_int(x), dpp8sel(1,0,3,2,5,4,7,6)));
}
__device__ __forceinline__ float dpp_xor2(float x) {
  return __int_as_float(__builtin_amdgcn_mov_dpp8(
      __float_as_int(x), dpp8sel(2,3,0,1,6,7,4,5)));
}
__device__ __forceinline__ float dpp_xor4(float x) {
  return __int_as_float(__builtin_amdgcn_mov_dpp8(
      __float_as_int(x), dpp8sel(4,5,6,7,0,1,2,3)));
}
__device__ __forceinline__ float dpp_xor8(float x) {
  int xi = __float_as_int(x);
  return __int_as_float(__builtin_amdgcn_update_dpp(
      xi, xi, 0x128 /*ROW_ROR:8*/, 0xf, 0xf, true));
}

__device__ __forceinline__ float rowmax16(float x) {
  x = fmaxf(x, dpp_xor1(x));
  x = fmaxf(x, dpp_xor2(x));
  x = fmaxf(x, dpp_xor4(x));
  x = fmaxf(x, dpp_xor8(x));
  return x;
}
__device__ __forceinline__ float rowsum16(float x) {
  x += dpp_xor1(x);
  x += dpp_xor2(x);
  x += dpp_xor4(x);
  x += dpp_xor8(x);
  return x;
}

// 16-bit WMMA A/B fragment: lane l holds row/col (l&15); K entries
// 8*(l>>4)+0..7 in elems 0..7 and 16+8*(l>>4)+0..7 in elems 8..15.
__device__ __forceinline__ v16h load_frag(const _Float16* base, int khalf) {
  v8h lo = *(const v8h*)(base + khalf * 8);
  v8h hi = *(const v8h*)(base + 16 + khalf * 8);
  return __builtin_shufflevector(lo, hi, 0,1,2,3,4,5,6,7,8,9,10,11,12,13,14,15);
}

__device__ __forceinline__ v8f wmma_f16(v16h a, v16h b, v8f c) {
  return __builtin_amdgcn_wmma_f32_16x16x32_f16(
      false, a, false, b, (short)0, c, false, false);
}

// ---- prep kernels --------------------------------------------------------

__global__ void gqa_cvt_f16(const float* __restrict__ in,
                            _Float16* __restrict__ out, int n, float scale) {
  int i = blockIdx.x * blockDim.x + threadIdx.x;
  if (i < n) out[i] = (_Float16)(in[i] * scale);
}

// v [B][S][KV*HD] f32 -> vt [B][KV][HD][S'] f16, s swizzled inside each
// 64-chunk: position p holds kv-local (p&3)*16 + (p>>2), so P stores pack
// into one b64 per row per lane while V loads stay contiguous.
__global__ void gqa_vtrans(const float* __restrict__ v,
                           _Float16* __restrict__ vt) {
  int idx = blockIdx.x * blockDim.x + threadIdx.x;
  if (idx >= BDIM * SEQ * KVE) return;
  int c  = idx & (KVE - 1);
  int bs = idx >> 9;
  int s  = bs & (SEQ - 1);
  int b  = bs >> 11;
  int h  = c >> 6;
  int d  = c & (HD - 1);
  int l  = s & 63;
  int sp = (s & ~63) | (((l & 15) << 2) | (l >> 4));
  vt[(((size_t)b * NKVH + h) * HD + d) * SEQ + sp] = (_Float16)v[idx];
}

// ---- flash attention: 8 waves/WG share K/V chunks via LDS ----------------
// Each wave owns one 16-row q-tile; WG covers 128 query rows of one q-head.

__global__ void __launch_bounds__(256)
gqa_attn(const _Float16* __restrict__ qh,   // [B][S][E] f16, pre-scaled
         const _Float16* __restrict__ kh,   // [B][S][KVE] f16
         const _Float16* __restrict__ vt,   // [B][KV][HD][S'] f16 (swizzled)
         float* __restrict__ out) {         // [B][S][E] f32
  __shared__ __align__(16) _Float16 kbuf[2][64 * 64];   // 2 x 8 KB
  __shared__ __align__(16) _Float16 vbuf[2][64 * 64];   // 2 x 8 KB
  __shared__ __align__(16) _Float16 pbuf[8][16 * 64];   // 8 x 2 KB

  const int tid   = threadIdx.x;
  const int wave  = tid >> 5;
  const int lane  = tid & 31;
  const int ln    = lane & 15;
  const int khalf = lane >> 4;
  const int h     = blockIdx.y;
  const int b     = blockIdx.z;
  const int kvh   = h >> 2;
  const int qtile = blockIdx.x * 8 + wave;

  const _Float16* Qb  = qh + ((size_t)b * SEQ + (size_t)qtile * 16) * EDIM + (size_t)h * HD;
  const _Float16* Kb  = kh + ((size_t)b * SEQ) * KVE + (size_t)kvh * HD;
  const _Float16* Vtb = vt + ((size_t)(b * NKVH + kvh) * HD) * SEQ;

  // cooperative-staging coordinates: 256 lanes x 32B cover one 8 KB buffer
  const int srow = tid >> 2;          // 0..63
  const int scol = (tid & 3) * 16;    // halfword offset, 32B per lane-row

  const _Float16* qrow = Qb + (size_t)ln * EDIM;
  v16h qa0 = load_frag(qrow, khalf);
  v16h qa1 = load_frag(qrow + 32, khalf);

  v8f acc[4];
  float m_i[8], lp[8];
#pragma unroll
  for (int t = 0; t < 4; ++t) { v8f z = {}; acc[t] = z; }
#pragma unroll
  for (int r = 0; r < 8; ++r) { m_i[r] = -INFINITY; lp[r] = 0.0f; }

  // ---- stage chunk 0
  {
    const v8h* ks = (const v8h*)(Kb + (size_t)srow * KVE + scol);
    const v8h* vs = (const v8h*)(Vtb + (size_t)srow * SEQ + scol);
    v8h k0 = ks[0], k1 = ks[1], v0 = vs[0], v1 = vs[1];
    *(v8h*)(&kbuf[0][srow * 64 + scol])     = k0;
    *(v8h*)(&kbuf[0][srow * 64 + scol + 8]) = k1;
    *(v8h*)(&vbuf[0][srow * 64 + scol])     = v0;
    *(v8h*)(&vbuf[0][srow * 64 + scol + 8]) = v1;
  }
  __syncthreads();

  _Float16* pl = &pbuf[wave][0];

  for (int kv0 = 0; kv0 < SEQ; kv0 += 64) {
    const int sel = (kv0 >> 6) & 1;
    const _Float16* kb = &kbuf[sel][0];
    const _Float16* vb = &vbuf[sel][0];
    const bool more = (kv0 + 64) < SEQ;

    // ---- issue next chunk's global loads early (hide latency)
    v8h kn0, kn1, vn0, vn1;
    if (more) {
      const v8h* ks = (const v8h*)(Kb + (size_t)(kv0 + 64 + srow) * KVE + scol);
      const v8h* vs = (const v8h*)(Vtb + (size_t)srow * SEQ + kv0 + 64 + scol);
      kn0 = ks[0]; kn1 = ks[1]; vn0 = vs[0]; vn1 = vs[1];
    }

    // ---- scores: Q(16x64) x K^T(64x64), operands from LDS (8 WMMA)
    v8f s[4];
#pragma unroll
    for (int t = 0; t < 4; ++t) {
      const _Float16* kr = kb + (t * 16 + ln) * 64;
      v8f z = {};
      s[t] = wmma_f16(qa0, load_frag(kr, khalf), z);
      s[t] = wmma_f16(qa1, load_frag(kr + 32, khalf), s[t]);
    }

    // ---- online softmax; scores already in exp2 domain (Q pre-scaled)
    float alpha[8];
#pragma unroll
    for (int r = 0; r < 8; ++r) {
      float a0 = s[0][r], a1 = s[1][r], a2 = s[2][r], a3 = s[3][r];
      float mx = rowmax16(fmaxf(fmaxf(a0, a1), fmaxf(a2, a3)));
      float nm = fmaxf(m_i[r], mx);
      float al = __builtin_amdgcn_exp2f(m_i[r] - nm);
      m_i[r] = nm;
      float e0 = __builtin_amdgcn_exp2f(a0 - nm);
      float e1 = __builtin_amdgcn_exp2f(a1 - nm);
      float e2 = __builtin_amdgcn_exp2f(a2 - nm);
      float e3 = __builtin_amdgcn_exp2f(a3 - nm);
      lp[r] = lp[r] * al + ((e0 + e1) + (e2 + e3));
      alpha[r] = al;
      int m = r + 8 * khalf;
      v4h pk = { (_Float16)e0, (_Float16)e1, (_Float16)e2, (_Float16)e3 };
      *(v4h*)(pl + m * 64 + 4 * ln) = pk;
    }
#pragma unroll
    for (int t = 0; t < 4; ++t)
#pragma unroll
      for (int r = 0; r < 8; ++r) acc[t][r] *= alpha[r];

    // P is wave-private; DS ops from one wave are in-order -> no s_barrier,
    // but stop the compiler from reordering the cross-lane store->load.
    __builtin_amdgcn_wave_barrier();
    v16h pa0 = load_frag(pl + ln * 64, khalf);
    v16h pa1 = load_frag(pl + ln * 64 + 32, khalf);

    // ---- P(16x64) x V(64x64), operands from LDS (8 WMMA)
#pragma unroll
    for (int t = 0; t < 4; ++t) {
      const _Float16* vr = vb + (t * 16 + ln) * 64;
      acc[t] = wmma_f16(pa0, load_frag(vr, khalf), acc[t]);
      acc[t] = wmma_f16(pa1, load_frag(vr + 32, khalf), acc[t]);
    }

    // ---- drain staged data into the other buffer, then sync the WG
    if (more) {
      _Float16* kd = &kbuf[sel ^ 1][0];
      _Float16* vd = &vbuf[sel ^ 1][0];
      *(v8h*)(kd + srow * 64 + scol)     = kn0;
      *(v8h*)(kd + srow * 64 + scol + 8) = kn1;
      *(v8h*)(vd + srow * 64 + scol)     = vn0;
      *(v8h*)(vd + srow * 64 + scol + 8) = vn1;
    }
    __syncthreads();
  }

  // ---- deferred row-sum reduction, normalize, store
  float inv[8];
#pragma unroll
  for (int r = 0; r < 8; ++r) inv[r] = 1.0f / rowsum16(lp[r]);

  float* ob = out + ((size_t)b * SEQ + (size_t)qtile * 16) * EDIM + (size_t)h * HD;
#pragma unroll
  for (int t = 0; t < 4; ++t)
#pragma unroll
    for (int r = 0; r < 8; ++r) {
      int m = r + 8 * khalf;
      ob[(size_t)m * EDIM + t * 16 + ln] = acc[t][r] * inv[r];
    }
}

// ---- launch --------------------------------------------------------------

extern "C" void kernel_launch(void* const* d_in, const int* in_sizes, int n_in,
                              void* d_out, int out_size, void* d_ws, size_t ws_size,
                              hipStream_t stream) {
  (void)in_sizes; (void)n_in; (void)out_size; (void)ws_size;
  const float* q = (const float*)d_in[0];
  const float* k = (const float*)d_in[1];
  const float* v = (const float*)d_in[2];
  float* out = (float*)d_out;

  _Float16* qh = (_Float16*)d_ws;                       // 16 MB
  _Float16* kh = qh + (size_t)BDIM * SEQ * EDIM;        //  4 MB
  _Float16* vt = kh + (size_t)BDIM * SEQ * KVE;         //  4 MB

  const int nq = BDIM * SEQ * EDIM;
  const int nk = BDIM * SEQ * KVE;

  gqa_cvt_f16<<<(nq + 255) / 256, 256, 0, stream>>>(q, qh, nq, QSCALE);
  gqa_cvt_f16<<<(nk + 255) / 256, 256, 0, stream>>>(k, kh, nk, 1.0f);
  gqa_vtrans <<<(nk + 255) / 256, 256, 0, stream>>>(v, vt);

  dim3 grid(SEQ / 128, NQH, BDIM);   // 16 x 32 x 2 = 1024 workgroups
  gqa_attn<<<grid, 256, 0, stream>>>(qh, kh, vt, out);
}
